// SWIDTBlock_6828998001253
// MI455X (gfx1250) — matrix operand order repaired and checked by reading, not compiled, not run on hardware
//
#include <hip/hip_runtime.h>
#include <hip/hip_bf16.h>

// ---------------------------------------------------------------------------
// Types for CDNA5 WMMA (wave32): bf16 16x16x32, f32 accumulate
// ---------------------------------------------------------------------------
typedef __bf16 bf16;
typedef __attribute__((ext_vector_type(16))) __bf16 v16bf;
typedef __attribute__((ext_vector_type(8)))  float  v8f;

union FragBf { v16bf v; uint4 u[2]; };

#define WMMA_BF16(a, b, c) \
  __builtin_amdgcn_wmma_f32_16x16x32_bf16(false, (a), false, (b), (short)0, (c), false, false)

static __device__ __forceinline__ v8f v8f_zero() {
  v8f z;
#pragma unroll
  for (int i = 0; i < 8; ++i) z[i] = 0.0f;
  return z;
}

static __device__ __forceinline__ float gelu_tanh(float x) {
  float x3 = x * x * x;
  return 0.5f * x * (1.0f + tanhf(0.7978845608028654f * (x + 0.044715f * x3)));
}

// VALU-only cross-lane shuffle within each row of 16 lanes (v_permlane16_b32).
static __device__ __forceinline__ float permlane16_f(float x, unsigned s1, unsigned s2) {
  unsigned u = __float_as_uint(x);
  unsigned r = __builtin_amdgcn_permlane16(u, u, s1, s2, false, false);
  return __uint_as_float(r);
}
// xor-butterfly max across the 16 lanes of each half-wave row
static __device__ __forceinline__ float rowmax16(float x) {
  x = fmaxf(x, permlane16_f(x, 0x67452301u, 0xEFCDAB89u));  // xor 1
  x = fmaxf(x, permlane16_f(x, 0x54761032u, 0xDCFE98BAu));  // xor 2
  x = fmaxf(x, permlane16_f(x, 0x32107654u, 0xBA98FEDCu));  // xor 4
  x = fmaxf(x, permlane16_f(x, 0xFEDCBA98u, 0x76543210u));  // xor 8
  return x;
}

// CDNA5 async global->LDS copy (16B per lane), tracked by ASYNCcnt.
static __device__ __forceinline__ void async_b128(unsigned lds_addr, const void* gptr) {
  unsigned long long ga = (unsigned long long)(uintptr_t)gptr;
  asm volatile("global_load_async_to_lds_b128 %0, %1, off"
               :: "v"(lds_addr), "v"(ga) : "memory");
}
static __device__ __forceinline__ void wait_async0() {
  asm volatile("s_wait_asynccnt 0x0" ::: "memory");
}
static __device__ __forceinline__ unsigned lds_addr_of(const void* p) {
  return (unsigned)(uintptr_t)p;  // LDS aperture: low 32 bits = LDS byte address
}

// ---------------------------------------------------------------------------
// Weight convert + transpose: in (K x N) f32 -> out (N x K) bf16
// ---------------------------------------------------------------------------
__global__ void transpose_to_bf16(const float* __restrict__ in, bf16* __restrict__ out,
                                  int K, int N) {
  size_t idx = (size_t)blockIdx.x * 256 + threadIdx.x;
  size_t total = (size_t)K * N;
  if (idx < total) {
    size_t n = idx / K;
    size_t k = idx % K;
    out[idx] = (bf16)in[k * (size_t)N + n];
  }
}

// V (B*T, D) bf16 -> Vt ((B*H*64), T) bf16  (head-dim-major so PV contraction is contiguous)
__global__ void v_transpose(const bf16* __restrict__ v, bf16* __restrict__ vt,
                            int B, int H, int T, int D) {
  size_t idx = (size_t)blockIdx.x * 256 + threadIdx.x;
  size_t total = (size_t)B * H * 64 * T;
  if (idx < total) {
    size_t t  = idx % T;
    size_t r  = idx / T;          // r = (b*H + h)*64 + d
    size_t d  = r & 63;
    size_t bh = r >> 6;
    size_t b  = bh >> 4;
    size_t h  = bh & 15;
    vt[idx] = v[(b * T + t) * (size_t)D + h * 64 + d];
  }
}

// ---------------------------------------------------------------------------
// LayerNorm: f32 in -> bf16 out, one 256-thread block per row
// ---------------------------------------------------------------------------
__global__ __launch_bounds__(256)
void layernorm_bf16(const float* __restrict__ x, const float* __restrict__ scale,
                    const float* __restrict__ bias, bf16* __restrict__ out, int D) {
  const int row = blockIdx.x;
  const float* xr = x + (size_t)row * D;
  float s = 0.f, ss = 0.f;
  for (int i = threadIdx.x; i < D; i += 256) {
    float v = xr[i];
    s += v;
    ss += v * v;
  }
#pragma unroll
  for (int d = 1; d < 32; d <<= 1) {
    s  += __shfl_xor(s, d, 32);
    ss += __shfl_xor(ss, d, 32);
  }
  __shared__ float red[2][8];
  const int wave = threadIdx.x >> 5, lane = threadIdx.x & 31;
  if (lane == 0) { red[0][wave] = s; red[1][wave] = ss; }
  __syncthreads();
  float S = 0.f, SS = 0.f;
#pragma unroll
  for (int w = 0; w < 8; ++w) { S += red[0][w]; SS += red[1][w]; }
  const float mu   = S / D;
  const float var  = SS / D - mu * mu;
  const float rstd = rsqrtf(var + 1e-6f);
  for (int i = threadIdx.x; i < D; i += 256) {
    out[(size_t)row * D + i] = (bf16)((xr[i] - mu) * rstd * scale[i] + bias[i]);
  }
}

// ---------------------------------------------------------------------------
// GEMM: C[M,N] = A[M,K](bf16) * Bt[N,K](bf16)^T + bias
//   MODE 0: store bf16 | MODE 1: gelu, store bf16 | MODE 2: + resid(f32), f32
// 128x128x64 tile, double-buffered LDS filled by async global->LDS (ASYNCcnt),
// 8 waves (2x4), wave tile 64x32, 16 WMMAs per k-step.
// ---------------------------------------------------------------------------
template <int MODE>
__global__ __launch_bounds__(256)
void gemm_bt(const bf16* __restrict__ A, const bf16* __restrict__ Bt,
             const float* __restrict__ bias, const float* __restrict__ resid,
             void* __restrict__ outp, int M, int N, int K) {
  __shared__ __align__(16) bf16 lA[2][128 * 64];
  __shared__ __align__(16) bf16 lB[2][128 * 64];

  const int tid   = threadIdx.x;
  const int lane  = tid & 31;
  const int wave  = tid >> 5;
  const int half  = lane >> 4;   // 0/1
  const int l16   = lane & 15;
  const int waveM = wave >> 2;   // 0..1
  const int waveN = wave & 3;    // 0..3
  const int bm    = blockIdx.y * 128;
  const int bn    = blockIdx.x * 128;

  // Per-thread staging assignment: 1024 16B-chunks per matrix, 4 per thread.
  const int r0 = tid >> 3;        // rows for c-th chunk: r0 + c*32
  const int c0 = (tid & 7) * 8;   // bf16 column offset of 16B chunk

  v8f acc[4][2];
#pragma unroll
  for (int i = 0; i < 4; ++i)
#pragma unroll
    for (int j = 0; j < 2; ++j) acc[i][j] = v8f_zero();

  const int nk = K >> 6;  // K / 64

  // Prologue: stage tile 0 into buffer 0.
#pragma unroll
  for (int c = 0; c < 4; ++c) {
    const int row = r0 + c * 32;
    async_b128(lds_addr_of(&lA[0][row * 64 + c0]), A  + (size_t)(bm + row) * K + c0);
    async_b128(lds_addr_of(&lB[0][row * 64 + c0]), Bt + (size_t)(bn + row) * K + c0);
  }
  wait_async0();
  __syncthreads();

  for (int t = 0; t < nk; ++t) {
    const int buf = t & 1;
    if (t + 1 < nk) {
      const int k1 = (t + 1) << 6;
#pragma unroll
      for (int c = 0; c < 4; ++c) {
        const int row = r0 + c * 32;
        async_b128(lds_addr_of(&lA[buf ^ 1][row * 64 + c0]),
                   A + (size_t)(bm + row) * K + k1 + c0);
        async_b128(lds_addr_of(&lB[buf ^ 1][row * 64 + c0]),
                   Bt + (size_t)(bn + row) * K + k1 + c0);
      }
    }

    const bf16* sA = lA[buf];
    const bf16* sB = lB[buf];
#pragma unroll
    for (int kk = 0; kk < 2; ++kk) {
      v16bf afrag[4], bfrag[2];
#pragma unroll
      for (int i = 0; i < 4; ++i) {
        const int m = waveM * 64 + i * 16 + l16;
        FragBf f;
        // CDNA5 16-bit A layout: lanes 0-15: K[0..7],[16..23]; lanes 16-31: K[8..15],[24..31]
        f.u[0] = *reinterpret_cast<const uint4*>(&sA[m * 64 + kk * 32 + half * 8]);
        f.u[1] = *reinterpret_cast<const uint4*>(&sA[m * 64 + kk * 32 + 16 + half * 8]);
        afrag[i] = f.v;
      }
#pragma unroll
      for (int j = 0; j < 2; ++j) {
        const int n = waveN * 32 + j * 16 + l16;
        FragBf f;
        // CDNA5 16-bit B layout: lanes 0-15: K[0..7],[16..23]; lanes 16-31: K[8..15],[24..31]
        f.u[0] = *reinterpret_cast<const uint4*>(&sB[n * 64 + kk * 32 + half * 8]);
        f.u[1] = *reinterpret_cast<const uint4*>(&sB[n * 64 + kk * 32 + 16 + half * 8]);
        bfrag[j] = f.v;
      }
#pragma unroll
      for (int i = 0; i < 4; ++i)
#pragma unroll
        for (int j = 0; j < 2; ++j)
          acc[i][j] = WMMA_BF16(afrag[i], bfrag[j], acc[i][j]);
    }

    wait_async0();      // our async fills of the other buffer are done
    __syncthreads();    // everyone's fills are done / readers finished
  }

  // Epilogue. C layout: VGPR v, lane -> M = v + 8*half, N = l16
#pragma unroll
  for (int i = 0; i < 4; ++i) {
#pragma unroll
    for (int j = 0; j < 2; ++j) {
      const int mBase = bm + waveM * 64 + i * 16 + half * 8;
      const int n     = bn + waveN * 32 + j * 16 + l16;
      const float bv  = bias ? bias[n] : 0.0f;
#pragma unroll
      for (int v = 0; v < 8; ++v) {
        const int m = mBase + v;
        float x = acc[i][j][v] + bv;
        if (MODE == 1) x = gelu_tanh(x);
        if (MODE == 2) {
          float* of = reinterpret_cast<float*>(outp);
          of[(size_t)m * N + n] = x + resid[(size_t)m * N + n];
        } else {
          bf16* ob = reinterpret_cast<bf16*>(outp);
          ob[(size_t)m * N + n] = (bf16)x;
        }
      }
    }
  }
}

// ---------------------------------------------------------------------------
// Flash attention (causal). 1 wave per 16-query tile, KV blocks of 32.
// Row max: v_permlane16 butterflies (VALU-only). Row sum: WMMA against ones.
// Main loop unmasked; masked tail handles the causal boundary blocks.
// ---------------------------------------------------------------------------
__global__ __launch_bounds__(128)
void flash_attn(const bf16* __restrict__ q, const bf16* __restrict__ k,
                const bf16* __restrict__ vt, bf16* __restrict__ o) {
  constexpr int T = 1024, D = 1024;
  __shared__ __align__(16) bf16 pstage[4 * 16 * 32];

  const int wave = threadIdx.x >> 5;
  const int lane = threadIdx.x & 31;
  const int half = lane >> 4;
  const int l16  = lane & 15;

  const int tile  = blockIdx.x * 4 + wave;
  const int bh    = tile >> 6;        // T/16 = 64 q-tiles per (b,h)
  const int qt    = tile & 63;
  const int b     = bh >> 4;
  const int h     = bh & 15;
  const int qbase = qt * 16;
  const int rowq  = b * T + qbase;

  bf16* ps = &pstage[wave * 16 * 32];

  // Q fragments (16x64 -> two 16x32 A-frags)
  v16bf qf[2];
#pragma unroll
  for (int c = 0; c < 2; ++c) {
    const bf16* qr = q + (size_t)(rowq + l16) * D + h * 64 + c * 32;
    FragBf f;
    f.u[0] = *reinterpret_cast<const uint4*>(qr + half * 16);
    f.u[1] = *reinterpret_cast<const uint4*>(qr + half * 16 + 8);
    qf[c] = f.v;
  }

  // all-ones B fragment for row-sum-by-WMMA
  v16bf ones;
#pragma unroll
  for (int i = 0; i < 16; ++i) ones[i] = (bf16)1.0f;

  v8f oacc[4];
#pragma unroll
  for (int t = 0; t < 4; ++t) oacc[t] = v8f_zero();
  float mrow[8], lrow[8];
#pragma unroll
  for (int v = 0; v < 8; ++v) { mrow[v] = -3.0e38f; lrow[v] = 0.0f; }

  // Hot-loop base pointers (per-lane), bumped by constant strides per block.
  const bf16* kcur = k  + (size_t)(b * T + l16) * D + h * 64 + half * 16;
  const bf16* vcur = vt + (size_t)(bh * 64 + l16) * T + half * 16;

  auto process_block = [&](int kb, bool masked) {
    // ---- load all 4 K fragments up front (clause-able, one wait)
    FragBf kf[4];
#pragma unroll
    for (int nt = 0; nt < 2; ++nt)
#pragma unroll
      for (int c = 0; c < 2; ++c) {
        const bf16* kp = kcur + nt * (16 * D) + c * 32;
        kf[nt * 2 + c].u[0] = *reinterpret_cast<const uint4*>(kp);
        kf[nt * 2 + c].u[1] = *reinterpret_cast<const uint4*>(kp + 8);
      }

    // ---- S = Q * K^T : two 16x16 tiles, contraction = head dim 64
    v8f s[2];
    s[0] = v8f_zero();
    s[1] = v8f_zero();
    s[0] = WMMA_BF16(qf[0], kf[0].v, s[0]);
    s[0] = WMMA_BF16(qf[1], kf[1].v, s[0]);
    s[1] = WMMA_BF16(qf[0], kf[2].v, s[1]);
    s[1] = WMMA_BF16(qf[1], kf[3].v, s[1]);

    // ---- issue V fragment loads now; latency overlaps the softmax VALU work
    FragBf vf[4];
#pragma unroll
    for (int t = 0; t < 4; ++t) {
      const bf16* vp = vcur + t * (16 * T);
      vf[t].u[0] = *reinterpret_cast<const uint4*>(vp);
      vf[t].u[1] = *reinterpret_cast<const uint4*>(vp + 8);
    }

    // ---- scale (+ causal mask on tail blocks) + online softmax (max only)
    float mnew[8];
#pragma unroll
    for (int v = 0; v < 8; ++v) {
      const int qrow_v = qbase + v + 8 * half;
#pragma unroll
      for (int nt = 0; nt < 2; ++nt) {
        float x = s[nt][v] * 0.125f;  // 1/sqrt(64)
        if (masked) {
          const int kcol = kb + nt * 16 + l16;
          if (kcol > qrow_v) x = -1.0e30f;
        }
        s[nt][v] = x;
      }
      const float mx = rowmax16(fmaxf(s[0][v], s[1][v]));
      mnew[v] = fmaxf(mrow[v], mx);
    }
#pragma unroll
    for (int v = 0; v < 8; ++v) {
      const float alpha = __expf(mrow[v] - mnew[v]);
      mrow[v] = mnew[v];
      lrow[v] *= alpha;
#pragma unroll
      for (int t = 0; t < 4; ++t) oacc[t][v] *= alpha;
    }
#pragma unroll
    for (int v = 0; v < 8; ++v) {
      s[0][v] = __expf(s[0][v] - mnew[v]);
      s[1][v] = __expf(s[1][v] - mnew[v]);
    }

    // ---- P (C-layout f32) -> LDS row-major 16x32 bf16 -> reload as A-frag
#pragma unroll
    for (int nt = 0; nt < 2; ++nt)
#pragma unroll
      for (int v = 0; v < 8; ++v)
        ps[(v + 8 * half) * 32 + nt * 16 + l16] = (bf16)s[nt][v];
    asm volatile("s_wait_dscnt 0" ::: "memory");  // wave-local LDS RAW

    v16bf pf;
    {
      FragBf f;
      f.u[0] = *reinterpret_cast<const uint4*>(&ps[l16 * 32 + half * 16]);
      f.u[1] = *reinterpret_cast<const uint4*>(&ps[l16 * 32 + half * 16 + 8]);
      pf = f.v;
    }

    // ---- row sums of P via WMMA against ones (replicated across lanes in C-layout)
    v8f lsum = WMMA_BF16(pf, ones, v8f_zero());
#pragma unroll
    for (int v = 0; v < 8; ++v) lrow[v] += lsum[v];

    // ---- O += P(16x32) * V(32x64): 4 output tiles, contraction = 32 KV rows
    oacc[0] = WMMA_BF16(pf, vf[0].v, oacc[0]);
    oacc[1] = WMMA_BF16(pf, vf[1].v, oacc[1]);
    oacc[2] = WMMA_BF16(pf, vf[2].v, oacc[2]);
    oacc[3] = WMMA_BF16(pf, vf[3].v, oacc[3]);

    kcur += 32 * D;  // next 32 KV rows
    vcur += 32;      // next 32 columns of Vt
  };

  // Unmasked main loop: whole 32-row block is below the causal diagonal
  // (kb+31 <= qbase  <=>  kb < qbase - 30).
  int kb = 0;
  for (; kb < qbase - 30; kb += 32) process_block(kb, false);
  // Masked tail (causal boundary).
  const int kmax = qbase + 16;
  for (; kb < kmax; kb += 32) process_block(kb, true);

  // ---- finalize: divide by l, store bf16
#pragma unroll
  for (int t = 0; t < 4; ++t) {
#pragma unroll
    for (int v = 0; v < 8; ++v) {
      const float val = oacc[t][v] / lrow[v];
      const int mr = rowq + v + 8 * half;
      const int col = h * 64 + t * 16 + l16;
      o[(size_t)mr * D + col] = (bf16)val;
    }
  }
}

// ---------------------------------------------------------------------------
// Launch
// ---------------------------------------------------------------------------
extern "C" void kernel_launch(void* const* d_in, const int* in_sizes, int n_in,
                              void* d_out, int out_size, void* d_ws, size_t ws_size,
                              hipStream_t stream) {
  (void)in_sizes; (void)n_in; (void)out_size; (void)ws_size;

  constexpr int B = 8, T = 1024, D = 1024, H = 16, DFF = 4096;
  constexpr int M = B * T;  // 8192

  const float* x    = (const float*)d_in[0];
  const float* ln1s = (const float*)d_in[2];
  const float* ln1b = (const float*)d_in[3];
  const float* wq   = (const float*)d_in[4];
  const float* bq   = (const float*)d_in[5];
  const float* wk   = (const float*)d_in[6];
  const float* bk   = (const float*)d_in[7];
  const float* wv   = (const float*)d_in[8];
  const float* bv   = (const float*)d_in[9];
  const float* wo   = (const float*)d_in[10];
  const float* bo   = (const float*)d_in[11];
  const float* ln2s = (const float*)d_in[12];
  const float* ln2b = (const float*)d_in[13];
  const float* w1   = (const float*)d_in[14];
  const float* b1   = (const float*)d_in[15];
  const float* w2   = (const float*)d_in[16];
  const float* b2   = (const float*)d_in[17];
  float* out = (float*)d_out;

  char* wsB = (char*)d_ws;
  size_t off = 0;
  auto alloc = [&](size_t bytes) -> void* {
    void* p = wsB + off;
    off += (bytes + 255) & ~(size_t)255;
    return p;
  };

  bf16* wqT = (bf16*)alloc((size_t)D * D * 2);
  bf16* wkT = (bf16*)alloc((size_t)D * D * 2);
  bf16* wvT = (bf16*)alloc((size_t)D * D * 2);
  bf16* woT = (bf16*)alloc((size_t)D * D * 2);
  bf16* w1T = (bf16*)alloc((size_t)D * DFF * 2);
  bf16* w2T = (bf16*)alloc((size_t)DFF * D * 2);
  bf16* hb  = (bf16*)alloc((size_t)M * D * 2);
  bf16* qb  = (bf16*)alloc((size_t)M * D * 2);
  bf16* kb  = (bf16*)alloc((size_t)M * D * 2);
  bf16* vb  = (bf16*)alloc((size_t)M * D * 2);
  bf16* vTb = (bf16*)alloc((size_t)M * D * 2);
  bf16* ob  = (bf16*)alloc((size_t)M * D * 2);
  float* x2 = (float*)alloc((size_t)M * D * 4);
  bf16* h2  = (bf16*)alloc((size_t)M * D * 2);
  bf16* f1  = (bf16*)alloc((size_t)M * DFF * 2);

  // 1) weight convert+transpose to bf16 (N x K)
  transpose_to_bf16<<<(D * D + 255) / 256, 256, 0, stream>>>(wq, wqT, D, D);
  transpose_to_bf16<<<(D * D + 255) / 256, 256, 0, stream>>>(wk, wkT, D, D);
  transpose_to_bf16<<<(D * D + 255) / 256, 256, 0, stream>>>(wv, wvT, D, D);
  transpose_to_bf16<<<(D * D + 255) / 256, 256, 0, stream>>>(wo, woT, D, D);
  transpose_to_bf16<<<(D * DFF + 255) / 256, 256, 0, stream>>>(w1, w1T, D, DFF);
  transpose_to_bf16<<<(DFF * D + 255) / 256, 256, 0, stream>>>(w2, w2T, DFF, D);

  // 2) LN1 -> bf16
  layernorm_bf16<<<M, 256, 0, stream>>>(x, ln1s, ln1b, hb, D);

  // 3) Q/K/V projections (WMMA, async-LDS staged)
  dim3 gD(D / 128, M / 128);       // (8, 64)
  gemm_bt<0><<<gD, 256, 0, stream>>>(hb, wqT, bq, nullptr, qb, M, D, D);
  gemm_bt<0><<<gD, 256, 0, stream>>>(hb, wkT, bk, nullptr, kb, M, D, D);
  gemm_bt<0><<<gD, 256, 0, stream>>>(hb, wvT, bv, nullptr, vb, M, D, D);

  // 4) V transpose for PV contraction
  v_transpose<<<(M * D + 255) / 256, 256, 0, stream>>>(vb, vTb, B, H, T, D);

  // 5) flash attention (WMMA): B*H*(T/16) = 8192 wave-tiles, 4 per block
  flash_attn<<<(B * H * (T / 16)) / 4, 128, 0, stream>>>(qb, kb, vTb, ob);

  // 6) output projection + residual (f32)
  gemm_bt<2><<<gD, 256, 0, stream>>>(ob, woT, bo, x, x2, M, D, D);

  // 7) LN2 -> bf16
  layernorm_bf16<<<M, 256, 0, stream>>>(x2, ln2s, ln2b, h2, D);

  // 8) FFN up + GELU
  dim3 gF1(DFF / 128, M / 128);    // (32, 64)
  gemm_bt<1><<<gF1, 256, 0, stream>>>(h2, w1T, b1, nullptr, f1, M, DFF, D);

  // 9) FFN down + bias + residual -> f32 output
  gemm_bt<2><<<gD, 256, 0, stream>>>(f1, w2T, b2, x2, out, M, D, DFF);
}
